// SparseAttention_53094385713648
// MI455X (gfx1250) — matrix operand order repaired and checked by reading, not compiled, not run on hardware
//
#include <hip/hip_runtime.h>

#define BB   2
#define SS   2048
#define DD   1024
#define HH   16
#define DKK  64
#define WIN  256

typedef __attribute__((ext_vector_type(16))) __bf16 v16bf;
typedef __attribute__((ext_vector_type(8)))  float  v8f;

union Frag { v16bf v; __bf16 h[16]; uint4 q[2]; };

__device__ __forceinline__ v8f wmma_bf16(v16bf a, v16bf b, v8f c) {
  return __builtin_amdgcn_wmma_f32_16x16x32_bf16(false, a, false, b, (short)0, c, false, false);
}

__device__ __forceinline__ v8f vzero8() {
  v8f r;
#pragma unroll
  for (int i = 0; i < 8; ++i) r[i] = 0.0f;
  return r;
}

__global__ void cvt_f32_to_bf16(const float* __restrict__ src, __bf16* __restrict__ dst, int n) {
  int i = blockIdx.x * blockDim.x + threadIdx.x;
  if (i < n) dst[i] = (__bf16)src[i];
}

// Y[M,N] = A[M,K](bf16) @ W[N,K]^T(bf16) + bias[N]
// mode 0: f32 out[m*N+n]
// mode 1: bf16 out[((b*H+h)*S+s)*DK+dk]   (Q,K layout)
// mode 2: bf16 out[((b*H+h)*DK+dk)*S+s]   (V transposed layout)
__global__ __launch_bounds__(256) void gemm_xwt(
    const __bf16* __restrict__ A, const __bf16* __restrict__ W,
    const float* __restrict__ bias, void* __restrict__ out,
    int M, int N, int K, int mode) {
  const int lane  = threadIdx.x & 31;
  const int wave  = threadIdx.x >> 5;
  const int half  = lane >> 4;
  const int col16 = lane & 15;
  const int mbase = blockIdx.y * 32 + (wave >> 2) * 16;
  const int nbase = blockIdx.x * 128 + (wave & 3) * 32;

  v8f acc0 = vzero8(), acc1 = vzero8();
  const __bf16* aptr  = A + (size_t)(mbase + col16) * K;
  const __bf16* wptr0 = W + (size_t)(nbase + col16) * K;
  const __bf16* wptr1 = W + (size_t)(nbase + 16 + col16) * K;

  for (int kc = 0; kc < K; kc += 32) {
    Frag a, b0, b1;
    a.q[0]  = *(const uint4*)(aptr + kc + half * 16);
    a.q[1]  = *(const uint4*)(aptr + kc + half * 16 + 8);
    const int ko = kc + half * 16;
    b0.q[0] = *(const uint4*)(wptr0 + ko);
    b0.q[1] = *(const uint4*)(wptr0 + ko + 8);
    b1.q[0] = *(const uint4*)(wptr1 + ko);
    b1.q[1] = *(const uint4*)(wptr1 + ko + 8);
    acc0 = wmma_bf16(a.v, b0.v, acc0);
    acc1 = wmma_bf16(a.v, b1.v, acc1);
  }

#pragma unroll
  for (int i = 0; i < 8; ++i) {
    int m = mbase + i + half * 8;
#pragma unroll
    for (int t = 0; t < 2; ++t) {
      int n = nbase + t * 16 + col16;
      float v = (t ? acc1[i] : acc0[i]) + bias[n];
      if (mode == 0) {
        ((float*)out)[(size_t)m * N + n] = v;
      } else {
        int b_ = m / SS, s_ = m % SS;
        int h_ = n / DKK, dk = n % DKK;
        size_t idx = (mode == 1)
            ? ((size_t)(b_ * HH + h_) * SS + s_) * DKK + dk
            : ((size_t)(b_ * HH + h_) * DKK + dk) * SS + s_;
        ((__bf16*)out)[idx] = (__bf16)v;
      }
    }
  }
}

// Flash attention over causal local window. One wave per 16-query tile.
// Scores computed transposed (S^T = K @ Q^T) so the softmax output registers
// are already in the A-fragment layout required for O = P @ V.
__global__ __launch_bounds__(256) void attn_swin(
    const __bf16* __restrict__ Qb, const __bf16* __restrict__ Kb,
    const __bf16* __restrict__ Vt, __bf16* __restrict__ Ob) {
  const int lane  = threadIdx.x & 31;
  const int wave  = threadIdx.x >> 5;
  const int half  = lane >> 4;
  const int col16 = lane & 15;
  const int gw    = blockIdx.x * 8 + wave;
  const int qt    = gw & (SS / 16 - 1);
  const int bh    = gw / (SS / 16);
  const int q0    = qt * 16;
  const int qg    = q0 + col16;   // this lane's query (as B-operand column)

  // Q fragments: B operand of K@Q^T, two dk-chunks of 32
  Frag qb0, qb1;
  {
    const __bf16* qp = Qb + ((size_t)bh * SS + qg) * DKK;
    qb0.q[0] = *(const uint4*)(qp + half * 8);
    qb0.q[1] = *(const uint4*)(qp + 16 + half * 8);
    qb1.q[0] = *(const uint4*)(qp + 32 + half * 8);
    qb1.q[1] = *(const uint4*)(qp + 48 + half * 8);
  }

  float m_run = -1e30f, l_run = 0.0f;
  v8f o[4];
#pragma unroll
  for (int t = 0; t < 4; ++t) o[t] = vzero8();

  int jlo = q0 - (WIN - 1); if (jlo < 0) jlo = 0;
  const int c0 = jlo >> 5;
  const int c1 = (q0 + 15) >> 5;

  for (int c = c0; c <= c1; ++c) {
    const int k0 = c * 32;

    // S^T tiles for the two 16-key groups of this 32-key chunk
    v8f st[2];
#pragma unroll
    for (int t = 0; t < 2; ++t) {
      const int keyrow = k0 + t * 16 + col16;
      const __bf16* kp = Kb + ((size_t)bh * SS + keyrow) * DKK;
      Frag ka0, ka1;
      ka0.q[0] = *(const uint4*)(kp + half * 8);
      ka0.q[1] = *(const uint4*)(kp + 16 + half * 8);
      ka1.q[0] = *(const uint4*)(kp + 32 + half * 8);
      ka1.q[1] = *(const uint4*)(kp + 48 + half * 8);
      v8f s = vzero8();
      s = wmma_bf16(ka0.v, qb0.v, s);
      s = wmma_bf16(ka1.v, qb1.v, s);
      st[t] = s;
    }

    // scale + causal-window mask; chunk row-max (per query = per lane pair)
    float mc = -1e30f;
#pragma unroll
    for (int i = 0; i < 8; ++i) {
      int j0 = k0 + i + half * 8;
      int j1 = j0 + 16;
      float s0 = (j0 <= qg && qg - j0 < WIN) ? st[0][i] * 0.125f : -1e30f;
      float s1 = (j1 <= qg && qg - j1 < WIN) ? st[1][i] * 0.125f : -1e30f;
      st[0][i] = s0; st[1][i] = s1;
      mc = fmaxf(mc, fmaxf(s0, s1));
    }
    mc = fmaxf(mc, __shfl_xor(mc, 16, 32));
    float m_new = fmaxf(m_run, mc);
    float alpha = __expf(m_run - m_new);

    float p0[8], p1[8], rs = 0.0f;
#pragma unroll
    for (int i = 0; i < 8; ++i) {
      float s0 = st[0][i];
      float s1 = st[1][i];
      p0[i] = (s0 > -1e29f) ? __expf(s0 - m_new) : 0.0f;
      p1[i] = (s1 > -1e29f) ? __expf(s1 - m_new) : 0.0f;
      rs += p0[i] + p1[i];
    }
    rs += __shfl_xor(rs, 16, 32);
    l_run = l_run * alpha + rs;
    m_run = m_new;

    // rescale O (O's C-layout has query = i + 8*half -> broadcast alpha)
    float av[8];
#pragma unroll
    for (int i = 0; i < 8; ++i) av[i] = __shfl(alpha, i + half * 8, 32);
#pragma unroll
    for (int i = 0; i < 8; ++i) {
      o[0][i] *= av[i]; o[1][i] *= av[i]; o[2][i] *= av[i]; o[3][i] *= av[i];
    }

    // P already in A-fragment order: pack to bf16
    Frag pf;
#pragma unroll
    for (int i = 0; i < 8; ++i) { pf.h[i] = (__bf16)p0[i]; pf.h[8 + i] = (__bf16)p1[i]; }

    // O += P @ V over 4 dk-column tiles; V stored [b,h,dk,s] so keys contiguous
#pragma unroll
    for (int t = 0; t < 4; ++t) {
      const int dk = t * 16 + col16;
      const __bf16* vp = Vt + ((size_t)bh * DKK + dk) * SS + k0 + half * 8;
      Frag vb;
      vb.q[0] = *(const uint4*)vp;
      vb.q[1] = *(const uint4*)(vp + 16);
      o[t] = wmma_bf16(pf.v, vb.v, o[t]);
    }
  }

  float inv = (l_run > 0.0f) ? 1.0f / l_run : 0.0f;
  float iv[8];
#pragma unroll
  for (int i = 0; i < 8; ++i) iv[i] = __shfl(inv, i + half * 8, 32);

  const int b_ = bh / HH, h_ = bh % HH;
#pragma unroll
  for (int t = 0; t < 4; ++t)
#pragma unroll
    for (int i = 0; i < 8; ++i) {
      int q = q0 + i + half * 8;
      int d = h_ * DKK + t * 16 + col16;
      Ob[(size_t)(b_ * SS + q) * DD + d] = (__bf16)(o[t][i] * iv[i]);
    }
}

extern "C" void kernel_launch(void* const* d_in, const int* in_sizes, int n_in,
                              void* d_out, int out_size, void* d_ws, size_t ws_size,
                              hipStream_t stream) {
  (void)in_sizes; (void)n_in; (void)out_size; (void)ws_size;
  const float* x  = (const float*)d_in[0];
  const float* wq = (const float*)d_in[1];
  const float* bq = (const float*)d_in[2];
  const float* wk = (const float*)d_in[3];
  const float* bk = (const float*)d_in[4];
  const float* wv = (const float*)d_in[5];
  const float* bv = (const float*)d_in[6];
  const float* wo = (const float*)d_in[7];
  const float* bo = (const float*)d_in[8];

  const size_t NX = (size_t)BB * SS * DD;   // 4,194,304
  const size_t NW = (size_t)DD * DD;        // 1,048,576
  char* ws = (char*)d_ws;
  __bf16* Xb  = (__bf16*)ws; ws += NX * 2;
  __bf16* Wqb = (__bf16*)ws; ws += NW * 2;
  __bf16* Wkb = (__bf16*)ws; ws += NW * 2;
  __bf16* Wvb = (__bf16*)ws; ws += NW * 2;
  __bf16* Wob = (__bf16*)ws; ws += NW * 2;
  __bf16* Qb  = (__bf16*)ws; ws += NX * 2;
  __bf16* Kb  = (__bf16*)ws; ws += NX * 2;
  __bf16* Vtb = (__bf16*)ws; ws += NX * 2;
  __bf16* Ob  = (__bf16*)ws; ws += NX * 2;

  cvt_f32_to_bf16<<<(unsigned)((NX + 255) / 256), 256, 0, stream>>>(x,  Xb,  (int)NX);
  cvt_f32_to_bf16<<<(unsigned)((NW + 255) / 256), 256, 0, stream>>>(wq, Wqb, (int)NW);
  cvt_f32_to_bf16<<<(unsigned)((NW + 255) / 256), 256, 0, stream>>>(wk, Wkb, (int)NW);
  cvt_f32_to_bf16<<<(unsigned)((NW + 255) / 256), 256, 0, stream>>>(wv, Wvb, (int)NW);
  cvt_f32_to_bf16<<<(unsigned)((NW + 255) / 256), 256, 0, stream>>>(wo, Wob, (int)NW);

  dim3 gg(DD / 128, (BB * SS) / 32);
  gemm_xwt<<<gg, 256, 0, stream>>>(Xb, Wqb, bq, Qb,  BB * SS, DD, DD, 1);
  gemm_xwt<<<gg, 256, 0, stream>>>(Xb, Wkb, bk, Kb,  BB * SS, DD, DD, 1);
  gemm_xwt<<<gg, 256, 0, stream>>>(Xb, Wvb, bv, Vtb, BB * SS, DD, DD, 2);

  attn_swin<<<(BB * HH * (SS / 16)) / 8, 256, 0, stream>>>(Qb, Kb, Vtb, Ob);

  gemm_xwt<<<gg, 256, 0, stream>>>(Ob, Wob, bo, d_out, BB * SS, DD, DD, 0);
}
